// MixedSparseTraditionalMLP_10608569221613
// MI455X (gfx1250) — matrix operand order repaired and checked by reading, not compiled, not run on hardware
//
#include <hip/hip_runtime.h>
#include <hip/hip_bf16.h>

// ---------------------------------------------------------------------------
// MixedSparseTraditionalMLP on MI455X (gfx1250, wave32, WMMA + TDM)
//
// Pipeline (all on `stream`, all scratch in d_ws):
//   1. convert_x:     x1 f32 [M,2048] -> xa_ext f16 [M,2080]  (col 2064=1.0, pad 0)
//   2. lora_t_f32:    t1 = x1 @ lora_a_up -> xa_ext cols [2048,2064)
//   3. dequant_ext:   w_up_q -> wup_ext f16 [H,2080] (+loraB cols, +bias col)
//   4. gemm<relu>:    x2_ext[M,8224] f16 = relu(xa_ext @ wup_ext^T)  (WMMA f16->f32)
//   5. lora_t_f16 + pad: t2 = x2 @ lora_a_down -> x2_ext cols [8192,8224)
//   6. dequant_ext:   w_down_q -> wdn_ext f16 [D,8224]
//   7. gemm<f32>:     out[M,2048] f32 = x2_ext @ wdn_ext^T
//
// Bias + LoRA are folded into the GEMMs by K-extension, so the WMMA loop does
// ALL the math.  GEMM tiles are staged into LDS by the Tensor Data Mover
// (tensor_load_to_lds + s_wait_tensorcnt) with a manual-staging fallback.
// ---------------------------------------------------------------------------

typedef __attribute__((ext_vector_type(16))) _Float16 v16h;
typedef __attribute__((ext_vector_type(8)))  _Float16 h8;
typedef __attribute__((ext_vector_type(8)))  float    v8f;
typedef __attribute__((ext_vector_type(4)))  float    f32x4;
typedef __attribute__((ext_vector_type(4)))  unsigned int u32x4;
typedef __attribute__((ext_vector_type(8)))  int i32x8;
typedef __attribute__((ext_vector_type(4)))  int i32x4;

union FragT { v16h h; u32x4 u[2]; };

static constexpr int M_  = 8192;   // B*S tokens
static constexpr int D_  = 2048;
static constexpr int H_  = 8192;
static constexpr int DE_ = 2080;   // D + 16 (lora) + 1 (bias), padded to K%32==0
static constexpr int HE_ = 8224;   // H + 32

#if defined(__has_builtin)
#if __has_builtin(__builtin_amdgcn_tensor_load_to_lds) && \
    __has_builtin(__builtin_amdgcn_s_wait_tensorcnt)
#define USE_TDM 1
#endif
#endif
#ifndef USE_TDM
#define USE_TDM 0
#endif

// ---------------------------------------------------------------------------
// 1. x1 f32 -> xa_ext f16, fill bias/pad columns (lora cols written by step 2)
// ---------------------------------------------------------------------------
__global__ __launch_bounds__(256) void convert_x_kernel(
    const float* __restrict__ x, _Float16* __restrict__ out)
{
    const int chunks = DE_ / 8;                       // 260 chunks of 8 halfs
    int idx = blockIdx.x * 256 + threadIdx.x;
    int m = idx / chunks, ck = idx % chunks;
    if (m >= M_) return;
    int k0 = ck * 8;
    if (k0 >= D_ && k0 < D_ + 16) return;             // lora region, filled later
    __align__(16) _Float16 o[8];
    if (k0 < D_) {
        const float* xp = x + (size_t)m * D_ + k0;
        #pragma unroll
        for (int i = 0; i < 8; ++i) o[i] = (_Float16)xp[i];
    } else {
        #pragma unroll
        for (int i = 0; i < 8; ++i) o[i] = (_Float16)((k0 + i == D_ + 16) ? 1.0f : 0.0f);
    }
    *(u32x4*)&out[(size_t)m * DE_ + k0] = *(const u32x4*)o;
}

// ---------------------------------------------------------------------------
// 2. t1[m,r] = sum_k x1[m,k] * loraA[k,r]  -> xa_ext[m, 2048+r]
//    all 32 lanes active (2 rows x 16 ranks per wave); X vectorized x4.
// ---------------------------------------------------------------------------
__global__ __launch_bounds__(256) void lora_t_f32_kernel(
    const float* __restrict__ X, const float* __restrict__ A,
    _Float16* __restrict__ out, int K, int Kext)
{
    int idx = blockIdx.x * 256 + threadIdx.x;
    int m = idx >> 4, r = idx & 15;
    if (m >= M_) return;
    const float* xp = X + (size_t)m * K;
    float s = 0.f;
    for (int k = 0; k < K; k += 4) {
        f32x4 xv = *(const f32x4*)(xp + k);
        #pragma unroll
        for (int i = 0; i < 4; ++i) s = fmaf(xv[i], A[(k + i) * 16 + r], s);
    }
    out[(size_t)m * Kext + K + r] = (_Float16)s;
}

// ---------------------------------------------------------------------------
// 5a. t2 from f16 activations (row stride Kext): X loads b128, A coalesced.
// ---------------------------------------------------------------------------
__global__ __launch_bounds__(256) void lora_t_f16_kernel(
    const _Float16* __restrict__ X, const float* __restrict__ A,
    _Float16* __restrict__ out, int K, int Kext)
{
    int idx = blockIdx.x * 256 + threadIdx.x;
    int m = idx >> 4, r = idx & 15;
    if (m >= M_) return;
    const _Float16* xp = X + (size_t)m * Kext;
    float s = 0.f;
    for (int k = 0; k < K; k += 8) {
        h8 xv = *(const h8*)(xp + k);
        #pragma unroll
        for (int i = 0; i < 8; ++i) s = fmaf((float)xv[i], A[(k + i) * 16 + r], s);
    }
    out[(size_t)m * Kext + K + r] = (_Float16)s;
}

// 5b. fill x2_ext cols [H+16, H+32): [1.0, 0, 0, ...]
__global__ __launch_bounds__(256) void pad_tail_kernel(_Float16* __restrict__ out)
{
    int m = blockIdx.x * 256 + threadIdx.x;
    if (m >= M_) return;
    __align__(16) _Float16 o[16];
    #pragma unroll
    for (int i = 0; i < 16; ++i) o[i] = (_Float16)(i == 0 ? 1.0f : 0.0f);
    _Float16* p = out + (size_t)m * HE_ + H_ + 16;
    *(u32x4*)p       = *(const u32x4*)&o[0];
    *(u32x4*)(p + 8) = *(const u32x4*)&o[8];
}

// ---------------------------------------------------------------------------
// 3/6. Dequant 4-bit codes (int32 storage) -> f16, K-extended with loraB rows
//      ([16,N] layout) and bias column.  out is [N,Kext] row-major == the
//      transposed-B (N-major) GEMM operand layout.
// ---------------------------------------------------------------------------
__global__ __launch_bounds__(256) void dequant_ext_kernel(
    const int* __restrict__ q, const float* __restrict__ scale,
    const float* __restrict__ loraB, const float* __restrict__ bias,
    _Float16* __restrict__ out, int N, int K, int Kext, int ldsc)
{
    const int chunks = Kext >> 3;
    int idx = blockIdx.x * 256 + threadIdx.x;
    int n = idx / chunks, ck = idx % chunks;
    if (n >= N) return;
    int k0 = ck * 8;
    __align__(16) _Float16 o[8];
    if (k0 < K) {
        float s = scale[(size_t)n * ldsc + (k0 >> 6)];   // 64-wide quant blocks
        const int* qp = q + (size_t)n * K + k0;
        #pragma unroll
        for (int i = 0; i < 8; ++i) o[i] = (_Float16)(((float)qp[i] - 7.5f) * s);
    } else {
        #pragma unroll
        for (int i = 0; i < 8; ++i) {
            int r = k0 + i - K;
            float v = (r < 16) ? loraB[(size_t)r * N + n]
                               : ((r == 16) ? bias[n] : 0.f);
            o[i] = (_Float16)v;
        }
    }
    *(u32x4*)&out[(size_t)n * Kext + k0] = *(const u32x4*)o;
}

#if USE_TDM
// ---------------------------------------------------------------------------
// Tensor Data Mover: DMA a 128-row x 32-half tile (row stride ldaElems) from
// global into LDS at ldsByteAddr.  D# packing per CDNA5 ISA §8.3/8.4:
//   group0: count=1 | lds_addr | global_addr[56:0] | type=2
//   group1: data_size=2B, tensor_dim0=32, tensor_dim1=128, tile_dim0=32,
//           tile_dim1=128, tensor_dim0_stride=ldaElems
// LDS fill order (X fastest, then Y) exactly matches the [128][32] tile.
// ---------------------------------------------------------------------------
__device__ __forceinline__ void tdm_load_tile_128x32(
    const _Float16* gsrc, unsigned ldaElems, unsigned ldsByteAddr)
{
    unsigned long long ga = (unsigned long long)(uintptr_t)gsrc;
    u32x4 g0;
    g0[0] = 1u;                                                 // count=1
    g0[1] = ldsByteAddr;                                        // lds_addr
    g0[2] = (unsigned)ga;                                       // global_addr lo
    g0[3] = (unsigned)((ga >> 32) & 0x01ffffffu) | (2u << 30);  // ga hi | type=2
    i32x8 g1;
    g1[0] = 0x00010000;          // workgroup_mask=0, data_size=1 (2 bytes)
    g1[1] = (int)(32u << 16);    // tensor_dim0 = 32   (bits [79:48], lo16)
    g1[2] = (int)(128u << 16);   // tensor_dim1 = 128  (bits [111:80], lo16)
    g1[3] = (int)(32u << 16);    // tile_dim0  = 32    (bits [127:112])
    g1[4] = 128;                 // tile_dim1  = 128   (bits [143:128])
    g1[5] = (int)ldaElems;       // tensor_dim0_stride[31:0]
    g1[6] = 0;                   // stride hi / dim1_stride lo
    g1[7] = 0;
    i32x4 z4 = {0, 0, 0, 0};
#if defined(__clang_major__) && __clang_major__ >= 23
    i32x8 z8 = {0, 0, 0, 0, 0, 0, 0, 0};
    __builtin_amdgcn_tensor_load_to_lds(g0, g1, z4, z4, z8, 0);
#else
    __builtin_amdgcn_tensor_load_to_lds(g0, g1, z4, z4, 0);
#endif
}
#endif  // USE_TDM

// ---------------------------------------------------------------------------
// WMMA GEMM:  C[M,N] = A[M,K] * Bt[N,K]^T   (f16 in, f32 accumulate)
// 256 threads / 8 waves, 128x128 tile, wave grid 4(M) x 2(N),
// per-wave 2x4 accumulators of v_wmma_f32_16x16x32_f16.
// LDS: A tile [128][32] + B tile (N-major) [128][32], double-buffered (32 KB),
// filled by TDM (wave 0 issues descriptors, s_wait_tensorcnt, one barrier/iter).
// ---------------------------------------------------------------------------
template <bool RELU_STORE_F16>
__global__ __launch_bounds__(256) void gemm_wmma_kernel(
    const _Float16* __restrict__ A, int lda,
    const _Float16* __restrict__ Bt, int ldb,
    void* __restrict__ Cout, int ldc, int K)
{
    __shared__ __align__(16) _Float16 As[2][128][32];
    __shared__ __align__(16) _Float16 Bs[2][128][32];

    const int tid  = threadIdx.x;
    const int lane = tid & 31;
    const int wave = tid >> 5;
    const int wm   = (wave & 3) * 32;   // wave M offset in tile
    const int wn   = (wave >> 2) * 64;  // wave N offset in tile
    const int m0   = blockIdx.y * 128;
    const int n0   = blockIdx.x * 128;
    const int g    = lane >> 4;         // half-lane group (ISA frag layout)
    const int l16  = lane & 15;

    const _Float16* Ab = A  + (size_t)m0 * lda;
    const _Float16* Bb = Bt + (size_t)n0 * ldb;

    v8f acc[2][4];
    #pragma unroll
    for (int i = 0; i < 2; ++i)
        #pragma unroll
        for (int j = 0; j < 4; ++j) acc[i][j] = v8f{};

#if USE_TDM
    const unsigned ldsA0 = (unsigned)(uintptr_t)&As[0][0][0];
    const unsigned ldsB0 = (unsigned)(uintptr_t)&Bs[0][0][0];
    if (wave == 0) {                       // one wave drives the DMA engine
        tdm_load_tile_128x32(Ab, (unsigned)lda, ldsA0);
        tdm_load_tile_128x32(Bb, (unsigned)ldb, ldsB0);
        __builtin_amdgcn_s_wait_tensorcnt(0);
    }
    __syncthreads();
#else
    // manual staging: 512 x 16B chunks per 8 KB tile, 2 per thread
    const int r0 = tid >> 2,         o0 = (tid & 3) * 8;
    const int r1 = (tid + 256) >> 2, o1 = (tid & 3) * 8;
    u32x4 ra0 = *(const u32x4*)(Ab + (size_t)r0 * lda + o0);
    u32x4 ra1 = *(const u32x4*)(Ab + (size_t)r1 * lda + o1);
    u32x4 rb0 = *(const u32x4*)(Bb + (size_t)r0 * ldb + o0);
    u32x4 rb1 = *(const u32x4*)(Bb + (size_t)r1 * ldb + o1);
    *(u32x4*)&As[0][r0][o0] = ra0;  *(u32x4*)&As[0][r1][o1] = ra1;
    *(u32x4*)&Bs[0][r0][o0] = rb0;  *(u32x4*)&Bs[0][r1][o1] = rb1;
    __syncthreads();
#endif

    const int KT = K >> 5;
    for (int kt = 0; kt < KT; ++kt) {
        const int cur = kt & 1;
        const bool more = (kt + 1) < KT;

#if USE_TDM
        if (more && wave == 0) {           // DMA next tiles while we compute
            const int kk = (kt + 1) << 5;
            tdm_load_tile_128x32(Ab + kk, (unsigned)lda, ldsA0 + (unsigned)(cur ^ 1) * 8192u);
            tdm_load_tile_128x32(Bb + kk, (unsigned)ldb, ldsB0 + (unsigned)(cur ^ 1) * 8192u);
        }
#else
        if (more) {
            const int kk = (kt + 1) << 5;
            ra0 = *(const u32x4*)(Ab + (size_t)r0 * lda + kk + o0);
            ra1 = *(const u32x4*)(Ab + (size_t)r1 * lda + kk + o1);
            rb0 = *(const u32x4*)(Bb + (size_t)r0 * ldb + kk + o0);
            rb1 = *(const u32x4*)(Bb + (size_t)r1 * ldb + kk + o1);
            if (kt + 2 < KT) {
                const int kp = (kt + 2) << 5;
                __builtin_prefetch(Ab + (size_t)r0 * lda + kp + o0, 0, 1);
                __builtin_prefetch(Bb + (size_t)r0 * ldb + kp + o0, 0, 1);
            }
        }
#endif

        // fragments per the 16-bit WMMA VGPR layouts:
        //  A (16x32): lane<16 -> M=lane, K 0..7 & 16..23 ; lane>=16 -> K 8..15 & 24..31
        //  B (32x16): lane<16 -> N=lane, K 0..7 & 16..23 ; lane>=16 -> K 8..15 & 24..31
        FragT af[2], bf[4];
        #pragma unroll
        for (int mf = 0; mf < 2; ++mf) {
            const int r = wm + mf * 16 + l16;
            af[mf].u[0] = *(const u32x4*)&As[cur][r][g * 8];
            af[mf].u[1] = *(const u32x4*)&As[cur][r][16 + g * 8];
        }
        #pragma unroll
        for (int nf = 0; nf < 4; ++nf) {
            const int c = wn + nf * 16 + l16;
            bf[nf].u[0] = *(const u32x4*)&Bs[cur][c][g * 8];
            bf[nf].u[1] = *(const u32x4*)&Bs[cur][c][16 + g * 8];
        }

        #pragma unroll
        for (int mf = 0; mf < 2; ++mf)
            #pragma unroll
            for (int nf = 0; nf < 4; ++nf)
                acc[mf][nf] = __builtin_amdgcn_wmma_f32_16x16x32_f16(
                    false, af[mf].h, false, bf[nf].h,
                    (short)0, acc[mf][nf], false, false);

#if USE_TDM
        if (more && wave == 0) __builtin_amdgcn_s_wait_tensorcnt(0);
        __syncthreads();
#else
        __syncthreads();
        if (more) {
            const int nxt = cur ^ 1;
            *(u32x4*)&As[nxt][r0][o0] = ra0;  *(u32x4*)&As[nxt][r1][o1] = ra1;
            *(u32x4*)&Bs[nxt][r0][o0] = rb0;  *(u32x4*)&Bs[nxt][r1][o1] = rb1;
            __syncthreads();
        }
#endif
    }

    // epilogue: C/D layout -> VGPR i holds (M = base + i + 8*g, N = base + l16)
    #pragma unroll
    for (int mf = 0; mf < 2; ++mf) {
        #pragma unroll
        for (int nf = 0; nf < 4; ++nf) {
            const int mb = m0 + wm + mf * 16 + g * 8;
            const int n  = n0 + wn + nf * 16 + l16;
            #pragma unroll
            for (int i = 0; i < 8; ++i) {
                const float v = acc[mf][nf][i];
                const size_t off = (size_t)(mb + i) * ldc + n;
                if constexpr (RELU_STORE_F16)
                    ((_Float16*)Cout)[off] = (_Float16)fmaxf(v, 0.0f);
                else
                    ((float*)Cout)[off] = v;
            }
        }
    }
}

// ---------------------------------------------------------------------------
extern "C" void kernel_launch(void* const* d_in, const int* in_sizes, int n_in,
                              void* d_out, int out_size, void* d_ws, size_t ws_size,
                              hipStream_t stream)
{
    (void)in_sizes; (void)n_in; (void)out_size; (void)ws_size;
    const float* x1    = (const float*)d_in[0];
    const int*   wupq  = (const int*)  d_in[1];
    const float* wupsc = (const float*)d_in[2];
    const float* bup   = (const float*)d_in[3];
    const float* lAu   = (const float*)d_in[4];
    const float* lBu   = (const float*)d_in[5];
    const int*   wdnq  = (const int*)  d_in[6];
    const float* wdnsc = (const float*)d_in[7];
    const float* bdn   = (const float*)d_in[8];
    const float* lAd   = (const float*)d_in[9];
    const float* lBd   = (const float*)d_in[10];
    float* out = (float*)d_out;

    char* ws = (char*)d_ws;
    const size_t oXA  = 0;
    const size_t oWUP = oXA  + (size_t)M_ * DE_ * 2;   //  34.1 MB
    const size_t oX2  = oWUP + (size_t)H_ * DE_ * 2;   //  68.2 MB
    const size_t oWDN = oX2  + (size_t)M_ * HE_ * 2;   // 202.9 MB
    _Float16* xa  = (_Float16*)(ws + oXA);
    _Float16* wup = (_Float16*)(ws + oWUP);
    _Float16* x2  = (_Float16*)(ws + oX2);
    _Float16* wdn = (_Float16*)(ws + oWDN);

    const dim3 blk(256);

    // 1. x1 -> f16 K-extended
    convert_x_kernel<<<(M_ * (DE_ / 8) + 255) / 256, blk, 0, stream>>>(x1, xa);
    // 2. t1 into xa_ext
    lora_t_f32_kernel<<<(M_ * 16 + 255) / 256, blk, 0, stream>>>(x1, lAu, xa, D_, DE_);
    // 3. dequant up weights, K-extended with loraB_up + bias
    dequant_ext_kernel<<<(H_ * (DE_ / 8) + 255) / 256, blk, 0, stream>>>(
        wupq, wupsc, lBu, bup, wup, H_, D_, DE_, D_ / 64);
    // 4. up GEMM + relu -> x2_ext f16
    gemm_wmma_kernel<true><<<dim3(H_ / 128, M_ / 128), blk, 0, stream>>>(
        xa, DE_, wup, DE_, x2, HE_, DE_);
    // 5. t2 + pad cols into x2_ext
    lora_t_f16_kernel<<<(M_ * 16 + 255) / 256, blk, 0, stream>>>(x2, lAd, x2, H_, HE_);
    pad_tail_kernel<<<(M_ + 255) / 256, blk, 0, stream>>>(x2);
    // 6. dequant down weights, K-extended with loraB_down + bias
    dequant_ext_kernel<<<(D_ * (HE_ / 8) + 255) / 256, blk, 0, stream>>>(
        wdnq, wdnsc, lBd, bdn, wdn, D_, H_, HE_, H_ / 64);
    // 7. down GEMM -> f32 output
    gemm_wmma_kernel<false><<<dim3(D_ / 128, M_ / 128), blk, 0, stream>>>(
        x2, HE_, wdn, HE_, out, D_, HE_);
}
